// Attention_41540923687523
// MI455X (gfx1250) — matrix operand order repaired and checked by reading, not compiled, not run on hardware
//
#include <hip/hip_runtime.h>
#include <hip/hip_bf16.h>

// ---------------------------------------------------------------------------
// Problem constants
// ---------------------------------------------------------------------------
#define EMBED   768
#define HEADS   12
#define DHEAD   64          // EMBED / HEADS
#define BATCH   32
#define SEQ     1024
#define TOKENS  (BATCH * SEQ)        // 32768
#define QKV_N   (3 * EMBED)          // 2304
#define HROW    (3 * DHEAD)          // 192 per-head row span inside a qkv row

// ---------------------------------------------------------------------------
// Types for WMMA fragments (CDNA5 wave32, v_wmma_f32_16x16x32_bf16)
// ---------------------------------------------------------------------------
typedef __attribute__((ext_vector_type(16))) __bf16 bf16x16;
typedef __attribute__((ext_vector_type(8)))  __bf16 bf16x8;
typedef __attribute__((ext_vector_type(8)))  float  f32x8;

// ---------------------------------------------------------------------------
// Fragment loaders (layouts per ISA 7.12.2, 16-bit A 16x32 / B 32x16)
// A (row-major [M][K], lda in elements):
//   lane<16 : row = lane,     elems 0..7 = K kb..kb+7,  elems 8..15 = kb+16..kb+23
//   lane>=16: row = lane-16,  kb = k0 + 8
// ---------------------------------------------------------------------------
__device__ __forceinline__ bf16x16
load_a_frag(const __bf16* __restrict__ A, size_t lda, int m0, int k0, int lane) {
    int row = m0 + (lane & 15);
    int kb  = k0 + ((lane >> 4) << 3);           // +8 for upper half-wave
    const __bf16* p = A + (size_t)row * lda + kb;
    bf16x8 lo = *(const bf16x8*)(p);
    bf16x8 hi = *(const bf16x8*)(p + 16);
    return __builtin_shufflevector(lo, hi, 0,1,2,3,4,5,6,7,8,9,10,11,12,13,14,15);
}

// B from a pre-transposed weight Bt (row-major [N][K], ldk in elements):
//   lane<16 : col = lane,     elems 0..7 = K k0..k0+7,  elems 8..15 = k0+16..k0+23
//   lane>=16: col = lane-16,  kb = k0 + 8
__device__ __forceinline__ bf16x16
load_b_frag(const __bf16* __restrict__ Bt, size_t ldk, int n0, int k0, int lane) {
    int col = n0 + (lane & 15);
    int kb  = k0 + ((lane >> 4) << 3);           // +8 for upper half-wave
    const __bf16* p = Bt + (size_t)col * ldk + kb;
    bf16x8 lo = *(const bf16x8*)(p);
    bf16x8 hi = *(const bf16x8*)(p + 16);
    return __builtin_shufflevector(lo, hi, 0,1,2,3,4,5,6,7,8,9,10,11,12,13,14,15);
}

// ---------------------------------------------------------------------------
// Conversion kernels (fp32 -> bf16; transpose for weights so the B-fragment
// becomes contiguous K per lane)
// ---------------------------------------------------------------------------
__global__ void cvt_f32_to_bf16(const float* __restrict__ in,
                                __bf16* __restrict__ out, size_t n) {
    size_t i = (size_t)blockIdx.x * blockDim.x + threadIdx.x;
    if (i < n) out[i] = (__bf16)in[i];
}

__global__ void transpose_cvt(const float* __restrict__ in,   // [K][N]
                              __bf16* __restrict__ out,       // [N][K]
                              int K, int N) {
    size_t i = (size_t)blockIdx.x * blockDim.x + threadIdx.x;
    if (i >= (size_t)K * N) return;
    int k = (int)(i / N);
    int n = (int)(i % N);
    out[(size_t)n * K + k] = (__bf16)in[i];
}

// ---------------------------------------------------------------------------
// Tiled WMMA GEMM:  C[M,N] = A[M,K] (bf16) * Bt[N,K]^T (bf16) + bias[N]
// One wave computes a (16*MT) x (16*NT) tile; fp32 accumulation.
// Each B fragment feeds MT WMMAs (operand reuse); next K-step prefetched
// via global_prefetch_b8.
// ---------------------------------------------------------------------------
template<bool OUT_BF16, int MT, int NT>
__global__ void gemm_wmma_bias(const __bf16* __restrict__ A,
                               const __bf16* __restrict__ Bt,
                               const float*  __restrict__ bias,
                               void* __restrict__ Cout,
                               int M, int N, int K) {
    const int lane   = threadIdx.x & 31;
    const int wave   = blockIdx.x * (blockDim.x >> 5) + (threadIdx.x >> 5);
    const int tilesN = N / (16 * NT);
    const int wm     = wave / tilesN;
    const int wn     = wave % tilesN;
    if (wm >= M / (16 * MT)) return;             // wave-uniform guard

    const int m0 = wm * (16 * MT);
    const int n0 = wn * (16 * NT);

    f32x8 acc[MT][NT] = {};

    for (int k0 = 0; k0 < K; k0 += 32) {
        // prefetch next K-step fragments (lowers to global_prefetch_b8)
        if (k0 + 32 < K) {
            const __bf16* pa = A + (size_t)(m0 + (lane & 15)) * K + (k0 + 32);
            __builtin_prefetch(pa, 0, 3);
            const __bf16* pb = Bt + (size_t)(n0 + (lane & 15)) * K + (k0 + 32);
            __builtin_prefetch(pb, 0, 3);
        }

        bf16x16 a[MT];
#pragma unroll
        for (int r = 0; r < MT; ++r)
            a[r] = load_a_frag(A, (size_t)K, m0 + r * 16, k0, lane);
#pragma unroll
        for (int t = 0; t < NT; ++t) {
            bf16x16 b = load_b_frag(Bt, (size_t)K, n0 + t * 16, k0, lane);
#pragma unroll
            for (int r = 0; r < MT; ++r)
                acc[r][t] = __builtin_amdgcn_wmma_f32_16x16x32_bf16(
                                false, a[r], false, b, (short)0, acc[r][t],
                                false, false);
        }
    }

    const int rowOff = (lane >> 4) << 3;         // +8 rows for upper half-wave
#pragma unroll
    for (int t = 0; t < NT; ++t) {
        int   col = n0 + t * 16 + (lane & 15);
        float bv  = bias[col];
#pragma unroll
        for (int r = 0; r < MT; ++r) {
#pragma unroll
            for (int e = 0; e < 8; ++e) {
                int   row = m0 + r * 16 + rowOff + e;
                float v   = acc[r][t][e] + bv;
                if (OUT_BF16) ((__bf16*)Cout)[(size_t)row * N + col] = (__bf16)v;
                else          ((float*) Cout)[(size_t)row * N + col] = v;
            }
        }
    }
}

// ---------------------------------------------------------------------------
// Per-token attention over heads.  One wave per token.
//  scores[h][g] = sum_d q[h][d] * k[g][d]   (two v_wmma_f32_16x16x32_bf16,
//                                            heads padded 12->16, clamped)
//  softmax over g (16-lane shfl_xor reductions, cols >= 12 masked to -inf)
//  ctx[h][d]    = sum_g attn[h][g] * v[g][d]  (fp32 VALU via LDS broadcast)
// ---------------------------------------------------------------------------
__global__ void attn_heads_kernel(const __bf16* __restrict__ qkv,  // [TOKENS][QKV_N]
                                  __bf16* __restrict__ ctx) {      // [TOKENS][EMBED]
    __shared__ float attn_s[8][16][17];          // per-wave 16x16 probs (padded)

    const int lane  = threadIdx.x & 31;
    const int w     = threadIdx.x >> 5;
    const int token = blockIdx.x * 8 + w;

    const __bf16* row = qkv + (size_t)token * QKV_N;

    // ---- scores = q @ k^T (padded to 16x16, K = 64 via two K=32 steps) ----
    int hc = (lane & 15); if (hc > HEADS - 1) hc = HEADS - 1;   // clamp pad rows
    f32x8 acc = {};
#pragma unroll
    for (int k0 = 0; k0 < DHEAD; k0 += 32) {
        // A-frag: q[h][k]  (h = clamped lane&15)
        int kbA = k0 + ((lane >> 4) << 3);
        const __bf16* pa = row + hc * HROW + kbA;               // q at +0
        bf16x8 alo = *(const bf16x8*)(pa);
        bf16x8 ahi = *(const bf16x8*)(pa + 16);
        bf16x16 a = __builtin_shufflevector(alo, ahi,
                     0,1,2,3,4,5,6,7,8,9,10,11,12,13,14,15);
        // B-frag: k^T  -> col g = clamped lane&15, same d order as A
        int kbB = k0 + ((lane >> 4) << 3);
        const __bf16* pb = row + hc * HROW + DHEAD + kbB;       // k at +64
        bf16x8 blo = *(const bf16x8*)(pb);
        bf16x8 bhi = *(const bf16x8*)(pb + 16);
        bf16x16 b = __builtin_shufflevector(blo, bhi,
                     0,1,2,3,4,5,6,7,8,9,10,11,12,13,14,15);
        acc = __builtin_amdgcn_wmma_f32_16x16x32_bf16(
                  false, a, false, b, (short)0, acc, false, false);
    }

    // ---- softmax over g (cols live across a 16-lane half-wave) ----
    const float scale = 0.03608439182435161f;    // 768^-0.5
    const bool padCol = (lane & 15) >= HEADS;
    const int  rowOff = (lane >> 4) << 3;
#pragma unroll
    for (int e = 0; e < 8; ++e) {
        float s = padCol ? -1e30f : acc[e] * scale;
        float m = s;
#pragma unroll
        for (int d = 1; d < 16; d <<= 1) m = fmaxf(m, __shfl_xor(m, d));
        float ex  = __expf(s - m);
        float sum = ex;
#pragma unroll
        for (int d = 1; d < 16; d <<= 1) sum += __shfl_xor(sum, d);
        attn_s[w][rowOff + e][lane & 15] = ex / sum;
    }

    // ---- ctx = attn @ v : lane owns columns d0 = lane, d1 = lane+32 ----
    float v0[HEADS], v1[HEADS];
#pragma unroll
    for (int g = 0; g < HEADS; ++g) {
        const __bf16* pv = row + g * HROW + 2 * DHEAD;          // v at +128
        v0[g] = (float)pv[lane];
        v1[g] = (float)pv[lane + 32];
    }
    __bf16* out = ctx + (size_t)token * EMBED;
#pragma unroll
    for (int h = 0; h < HEADS; ++h) {
        float s0 = 0.f, s1 = 0.f;
#pragma unroll
        for (int g = 0; g < HEADS; ++g) {
            float p = attn_s[w][h][g];
            s0 = fmaf(p, v0[g], s0);
            s1 = fmaf(p, v1[g], s1);
        }
        out[h * DHEAD + lane]      = (__bf16)s0;
        out[h * DHEAD + lane + 32] = (__bf16)s1;
    }
}

// ---------------------------------------------------------------------------
// Workspace layout (bytes), all 4 KB aligned
// ---------------------------------------------------------------------------
#define OFF_XBF   ((size_t)0)                                   // 48 MB
#define OFF_WQKVT (OFF_XBF   + (size_t)TOKENS * EMBED * 2)      // 3.375 MB
#define OFF_WOT   (OFF_WQKVT + (size_t)EMBED * QKV_N * 2)       // 1.125 MB
#define OFF_CTX   (OFF_WOT   + (size_t)EMBED * EMBED * 2)       // 48 MB
#define OFF_QKV   (OFF_CTX   + (size_t)TOKENS * EMBED * 2)      // 144 MB

extern "C" void kernel_launch(void* const* d_in, const int* in_sizes, int n_in,
                              void* d_out, int out_size, void* d_ws, size_t ws_size,
                              hipStream_t stream) {
    const float* x     = (const float*)d_in[0];
    const float* w_qkv = (const float*)d_in[1];
    const float* b_qkv = (const float*)d_in[2];
    const float* w_o   = (const float*)d_in[3];
    const float* b_o   = (const float*)d_in[4];
    float* out = (float*)d_out;

    char* ws = (char*)d_ws;
    __bf16* x_bf    = (__bf16*)(ws + OFF_XBF);
    __bf16* wqkvT   = (__bf16*)(ws + OFF_WQKVT);
    __bf16* woT     = (__bf16*)(ws + OFF_WOT);
    __bf16* ctx_bf  = (__bf16*)(ws + OFF_CTX);
    __bf16* qkv_bf  = (__bf16*)(ws + OFF_QKV);

    // 1) precision / layout conversion
    {
        size_t n = (size_t)TOKENS * EMBED;                      // 25,165,824
        cvt_f32_to_bf16<<<(unsigned)(n / 256), 256, 0, stream>>>(x, x_bf, n);
        size_t nq = (size_t)EMBED * QKV_N;                      // 1,769,472
        transpose_cvt<<<(unsigned)((nq + 255) / 256), 256, 0, stream>>>(
            w_qkv, wqkvT, EMBED, QKV_N);
        size_t no = (size_t)EMBED * EMBED;                      // 589,824
        transpose_cvt<<<(unsigned)((no + 255) / 256), 256, 0, stream>>>(
            w_o, woT, EMBED, EMBED);
    }

    // 2) qkv = x @ w_qkv + b_qkv   (bf16 out), 32x64 wave tiles
    {
        int waves  = (TOKENS / 32) * (QKV_N / 64);              // 36,864
        int blocks = waves / 8;                                 // 4,608
        gemm_wmma_bias<true, 2, 4><<<blocks, 256, 0, stream>>>(
            x_bf, wqkvT, b_qkv, (void*)qkv_bf, TOKENS, QKV_N, EMBED);
    }

    // 3) per-token attention over heads
    attn_heads_kernel<<<TOKENS / 8, 256, 0, stream>>>(qkv_bf, ctx_bf);

    // 4) out = ctx @ w_o + b_o     (fp32 out), 32x64 wave tiles
    {
        int waves  = (TOKENS / 32) * (EMBED / 64);              // 12,288
        int blocks = waves / 8;                                 // 1,536
        gemm_wmma_bias<false, 2, 4><<<blocks, 256, 0, stream>>>(
            ctx_bf, woT, b_o, (void*)out, TOKENS, EMBED, EMBED);
    }
}
